// Conv_KNRM_69028714381699
// MI455X (gfx1250) — matrix operand order repaired and checked by reading, not compiled, not run on hardware
//
#include <hip/hip_runtime.h>
#include <hip/hip_bf16.h>

typedef __attribute__((ext_vector_type(16))) _Float16 halfx16;
typedef __attribute__((ext_vector_type(8)))  _Float16 halfx8;
typedef __attribute__((ext_vector_type(8)))  float    floatx8;

#define B_   128
#define Q_   30
#define D_   200
#define E_   300
#define C_   128
#define QPAD 32     // 2 M-tiles of 16
#define DPAD 224    // 13 N-tiles of 16 used (208), padded to chunk multiple of 32

// ---------------------------------------------------------------------------
// Stage 0: masks (padded with zeros so WMMA padding rows contribute nothing)
// ---------------------------------------------------------------------------
__global__ __launch_bounds__(256)
void knrm_mask_kernel(const int* __restrict__ qtok, const int* __restrict__ dtok,
                      float* __restrict__ qm, float* __restrict__ dm) {
    int b = blockIdx.x, t = threadIdx.x;
    if (t < QPAD)
        qm[b * QPAD + t] = (t < Q_ && qtok[b * Q_ + t] > 0) ? 1.f : 0.f;
    int i = t - QPAD;
    if (i >= 0 && i < 208)
        dm[b * 208 + i] = (i < D_ && dtok[b * D_ + i] > 0) ? 1.f : 0.f;
}

// ---------------------------------------------------------------------------
// Stage 1: embedding gather + n-gram conv + ReLU + L2-normalize -> f16
// grid = (B, 6): y/3 = side (0=query,1=doc), y%3 = gram (k = g+1)
// block = 128 threads, thread = output channel
// ---------------------------------------------------------------------------
__global__ __launch_bounds__(128)
void knrm_conv_kernel(const int* __restrict__ qtok, const int* __restrict__ dtok,
                      const float* __restrict__ emb,
                      const float* __restrict__ w1, const float* __restrict__ w2,
                      const float* __restrict__ w3,
                      const float* __restrict__ b1, const float* __restrict__ b2,
                      const float* __restrict__ b3,
                      _Float16* __restrict__ Qn, _Float16* __restrict__ Dn) {
    int b = blockIdx.x;
    int v = blockIdx.y;
    int side = v / 3;
    int g = v % 3;
    int k = g + 1;
    int L    = side ? D_ : Q_;
    int Lpad = side ? DPAD : QPAD;
    const int*   tok  = side ? (dtok + b * D_) : (qtok + b * Q_);
    const float* w    = (g == 0) ? w1 : (g == 1 ? w2 : w3);
    const float* bias = (g == 0) ? b1 : (g == 1 ? b2 : b3);
    _Float16* out = side ? (Dn + (size_t)(b * 3 + g) * DPAD * C_)
                         : (Qn + (size_t)(b * 3 + g) * QPAD * C_);
    int c = threadIdx.x;

    __shared__ float xs[34 * E_];   // embedding window (32 + k - 1 rows)
    __shared__ float ys[32 * C_];   // relu'd conv outputs
    __shared__ float nrm[32];       // 1/(||y|| + eps)

    for (int l0 = 0; l0 < Lpad; l0 += 32) {
        int nrows = 32 + k - 1;
        __syncthreads();
        // cooperative gather of embedding rows (zero outside real length)
        for (int row = 0; row < nrows; ++row) {
            int l = l0 + row;
            if (l < L) {
                const float* er = emb + (size_t)tok[l] * E_;
                for (int e = c; e < E_; e += 128) xs[row * E_ + e] = er[e];
            } else {
                for (int e = c; e < E_; e += 128) xs[row * E_ + e] = 0.f;
            }
        }
        __syncthreads();

        float acc[32];
        float bv = bias[c];
#pragma unroll
        for (int l = 0; l < 32; ++l) acc[l] = bv;
        const float* wc = w + (size_t)c * E_ * k;   // w[c][e][t]
        for (int e = 0; e < E_; ++e) {
            for (int t = 0; t < k; ++t) {
                float wv = wc[e * k + t];
#pragma unroll
                for (int l = 0; l < 32; ++l)
                    acc[l] += xs[(l + t) * E_ + e] * wv;  // LDS broadcast reads
            }
        }
#pragma unroll
        for (int l = 0; l < 32; ++l) ys[l * C_ + c] = fmaxf(acc[l], 0.f);
        __syncthreads();
        if (c < 32) {
            float s = 0.f;
            for (int j = 0; j < C_; ++j) { float y = ys[c * C_ + j]; s += y * y; }
            nrm[c] = 1.f / (sqrtf(s) + 1e-13f);
        }
        __syncthreads();
#pragma unroll
        for (int l = 0; l < 32; ++l) {
            int gl = l0 + l;
            float val = (gl < L) ? ys[l * C_ + c] * nrm[l] : 0.f;
            out[(size_t)gl * C_ + c] = (_Float16)val;
        }
    }
}

// ---------------------------------------------------------------------------
// Stage 2: cosine GEMM (WMMA f16 -> f32) + Gaussian soft-bin pooling.
// grid = (9 pairs, B); block = 128 (4 waves).
// Qn: [b][g][32][128] (M x K), Dn: [b][g][224][128] (N x K).
// Doc-dim reduction is accumulated PER LANE in registers across N-tiles;
// a single ds_add_f32 per (q,kernel) merges at the end (no bpermute chains).
// ---------------------------------------------------------------------------
__global__ __launch_bounds__(128)
void knrm_pool_kernel(const _Float16* __restrict__ Qn, const _Float16* __restrict__ Dn,
                      const float* __restrict__ qmaskf, const float* __restrict__ dmaskf,
                      float* __restrict__ feats) {
    __shared__ float pkq[32][12];   // per (q, kernel) accumulator
    int tid  = threadIdx.x;
    int wave = tid >> 5;
    int lane = tid & 31;
    int hs   = lane >> 4;     // half-wave select
    int nl   = lane & 15;
    int b    = blockIdx.y;
    int pair = blockIdx.x;
    int qg = pair / 3, dg = pair % 3;

    for (int i = tid; i < 32 * 12; i += 128) ((float*)pkq)[i] = 0.f;
    __syncthreads();

    const _Float16* Abase = Qn + (size_t)(b * 3 + qg) * QPAD * C_;
    const _Float16* Bbase = Dn + (size_t)(b * 3 + dg) * DPAD * C_;

    const float mu[11]  = {1.0f, 0.9f, 0.7f, 0.5f, 0.3f, 0.1f,
                           -0.1f, -0.3f, -0.5f, -0.7f, -0.9f};
    const float i2s[11] = {500000.f, 50.f, 50.f, 50.f, 50.f, 50.f,
                           50.f, 50.f, 50.f, 50.f, 50.f};

    for (int mt = 0; mt < 2; ++mt) {
        // A fragments for this M-tile: lane nl = row M, half-wave selects K block
        halfx16 afrag[4];
        const _Float16* arow = Abase + (size_t)(mt * 16 + nl) * C_;
#pragma unroll
        for (int kb = 0; kb < 4; ++kb) {
            int k0 = kb * 32 + hs * 8;
            ((halfx8*)&afrag[kb])[0] = *(const halfx8*)(arow + k0);
            ((halfx8*)&afrag[kb])[1] = *(const halfx8*)(arow + k0 + 16);
        }
        float qmv[8];
#pragma unroll
        for (int r = 0; r < 8; ++r)
            qmv[r] = qmaskf[b * QPAD + mt * 16 + hs * 8 + r];

        float acc[8][11];
#pragma unroll
        for (int r = 0; r < 8; ++r)
#pragma unroll
            for (int kk = 0; kk < 11; ++kk) acc[r][kk] = 0.f;

        for (int nt = wave; nt < 13; nt += 4) {
            const _Float16* brow = Bbase + (size_t)(nt * 16 + nl) * C_;
            halfx16 bfrag[4];
#pragma unroll
            for (int kb = 0; kb < 4; ++kb) {
                int k0 = kb * 32 + hs * 8;
                ((halfx8*)&bfrag[kb])[0] = *(const halfx8*)(brow + k0);
                ((halfx8*)&bfrag[kb])[1] = *(const halfx8*)(brow + k0 + 16);
            }
            float dmv = dmaskf[b * 208 + nt * 16 + nl];

            floatx8 c = {0.f, 0.f, 0.f, 0.f, 0.f, 0.f, 0.f, 0.f};
#pragma unroll
            for (int kb = 0; kb < 4; ++kb)
                c = __builtin_amdgcn_wmma_f32_16x16x32_f16(
                        false, afrag[kb], false, bfrag[kb],
                        (short)0, c, false, false);
#pragma unroll
            for (int r = 0; r < 8; ++r) {
                float m = qmv[r] * dmv;
                float s = c[r] * m;
#pragma unroll
                for (int kk = 0; kk < 11; ++kk) {
                    float d = s - mu[kk];
                    acc[r][kk] += __expf(-(d * d) * i2s[kk]) * m;
                }
            }
        }
        // merge per-lane partials (fire-and-forget LDS atomics, no waits)
#pragma unroll
        for (int r = 0; r < 8; ++r) {
            int q = mt * 16 + hs * 8 + r;
#pragma unroll
            for (int kk = 0; kk < 11; ++kk)
                atomicAdd(&pkq[q][kk], acc[r][kk]);
        }
    }
    __syncthreads();
    if (tid < 11) {
        float a = 0.f;
        for (int q = 0; q < Q_; ++q) {
            float lv = __logf(fmaxf(pkq[q][tid], 1e-10f)) * 0.01f;
            a += lv * qmaskf[b * QPAD + q];
        }
        feats[b * 99 + pair * 11 + tid] = a;
    }
}

// ---------------------------------------------------------------------------
// Stage 3: final dense projection -> [B]
// ---------------------------------------------------------------------------
__global__ __launch_bounds__(128)
void knrm_dense_kernel(const float* __restrict__ feats, const float* __restrict__ dw,
                       float* __restrict__ outv) {
    int b = threadIdx.x;
    float a = 0.f;
    for (int j = 0; j < 99; ++j) a += feats[b * 99 + j] * dw[j];
    outv[b] = a;
}

extern "C" void kernel_launch(void* const* d_in, const int* in_sizes, int n_in,
                              void* d_out, int out_size, void* d_ws, size_t ws_size,
                              hipStream_t stream) {
    const int*   qtok = (const int*)d_in[0];
    const int*   dtok = (const int*)d_in[1];
    const float* emb  = (const float*)d_in[2];
    const float* w1   = (const float*)d_in[3];
    const float* w2   = (const float*)d_in[4];
    const float* w3   = (const float*)d_in[5];
    const float* b1   = (const float*)d_in[6];
    const float* b2   = (const float*)d_in[7];
    const float* b3   = (const float*)d_in[8];
    const float* dw   = (const float*)d_in[9];

    char* ws = (char*)d_ws;
    size_t off = 0;
    _Float16* Qn = (_Float16*)(ws + off); off += (size_t)3 * B_ * QPAD * C_ * sizeof(_Float16);
    _Float16* Dn = (_Float16*)(ws + off); off += (size_t)3 * B_ * DPAD * C_ * sizeof(_Float16);
    float* qm    = (float*)(ws + off);    off += (size_t)B_ * QPAD * sizeof(float);
    float* dm    = (float*)(ws + off);    off += (size_t)B_ * 208 * sizeof(float);
    float* feats = (float*)(ws + off);    off += (size_t)B_ * 99 * sizeof(float);

    knrm_mask_kernel<<<B_, 256, 0, stream>>>(qtok, dtok, qm, dm);
    knrm_conv_kernel<<<dim3(B_, 6), 128, 0, stream>>>(qtok, dtok, emb,
                                                      w1, w2, w3, b1, b2, b3, Qn, Dn);
    knrm_pool_kernel<<<dim3(9, B_), 128, 0, stream>>>(Qn, Dn, qm, dm, feats);
    knrm_dense_kernel<<<1, 128, 0, stream>>>(feats, dw, (float*)d_out);
}
